// GCNNet_38036230373710
// MI455X (gfx1250) — compile-verified, matrix operand order checked by reading, not run on hardware
//
#include <hip/hip_runtime.h>
#include <hip/hip_bf16.h>
#include <math.h>

// ---------------------------------------------------------------------------
// CDNA5 (gfx1250) wave32 WMMA types
// ---------------------------------------------------------------------------
typedef __attribute__((ext_vector_type(16))) _Float16 v16h;
typedef __attribute__((ext_vector_type(8)))  float    v8f;

// Order-preserving float <-> uint key (for atomic max on floats)
__device__ __forceinline__ unsigned ford(float f) {
    unsigned u = __float_as_uint(f);
    return (u & 0x80000000u) ? ~u : (u | 0x80000000u);
}
__device__ __forceinline__ float fdec(unsigned k) {
    unsigned u = (k & 0x80000000u) ? (k ^ 0x80000000u) : ~k;
    return __uint_as_float(u);
}

// K index inside a 16-bit 16x32 WMMA A/B fragment for half-slot h (0..15)
// (CDNA5 ISA: lanes 0-15 hold K 0-7,16-23; lanes 16-31 hold K 8-15,24-31)
__device__ __forceinline__ int frag_k(int h, int lane) {
    return (h < 8 ? h : h + 8) + ((lane & 16) ? 8 : 0);
}

// ---------------------------------------------------------------------------
// Pack weight matrix B[K,Nc] (f32, row-major) into f16 WMMA fragment order:
//   Bp[((tn*kchunks + chunk)*32 + lane)*16 + h] = B[chunk*32 + frag_k(h,lane)][tn*16 + lane%16]
// One thread per (tn, chunk, lane); unconditional loads, value-masked tail.
// ---------------------------------------------------------------------------
__global__ void pack_b_kernel(const float* __restrict__ B,
                              _Float16* __restrict__ Bp,
                              int K, int Nc) {
    const int kchunks = K >> 5;
    const int tilesN  = (Nc + 15) >> 4;
    const int total   = tilesN * kchunks * 32;
    int t = blockIdx.x * blockDim.x + threadIdx.x;
    if (t >= total) return;
    const int lane  = t & 31;
    const int rest  = t >> 5;
    const int chunk = rest % kchunks;
    const int tn    = rest / kchunks;
    const int col   = tn * 16 + (lane & 15);
    const bool cok  = col < Nc;
    const int colc  = cok ? col : 0;
    _Float16* dst = Bp + (size_t)t * 16;
    #pragma unroll
    for (int h = 0; h < 16; ++h) {
        int kk  = (chunk << 5) + frag_k(h, lane);
        float v = B[(size_t)kk * Nc + colc];
        dst[h]  = cok ? (_Float16)v : (_Float16)0.0f;
    }
}

// ---------------------------------------------------------------------------
// WMMA GEMM: C[M,Nc] = act(A[M,K] @ B + bias), B pre-packed to fragments.
//   One wave32 per 16x16 output tile; K multiple of 32 (true for all layers).
//   A-fragment: 4x global_load_b128 per K-chunk (contiguous K runs).
//   B-fragment: one aligned v16h (32B) load per K-chunk.
//   act: 0=none 1=relu 2=sigmoid 3=relu->sigmoid
// ---------------------------------------------------------------------------
__global__ void wmma_gemm_kernel(const float* __restrict__ A,
                                 const v16h* __restrict__ Bp,
                                 const float* __restrict__ bias,
                                 float* __restrict__ C,
                                 int M, int Nc, int K, int act) {
    const int lane   = threadIdx.x & 31;
    const int tilesN = (Nc + 15) >> 4;
    const int tilesM = (M + 15) >> 4;
    const int tile   = blockIdx.x * blockDim.y + threadIdx.y;
    if (tile >= tilesM * tilesN) return;
    const int tm = tile / tilesN;
    const int tn = tile - tm * tilesN;
    const int m0 = tm << 4;
    const int n0 = tn << 4;

    const int r     = lane & 15;
    const int row   = m0 + r;
    const int row_c = row < M ? row : (M - 1);   // clamp: garbage rows never stored
    const int q     = (lane & 16) ? 8 : 0;       // K sub-offset for this half-wave

    const int kchunks = K >> 5;
    const float* Abase = A + (size_t)row_c * K;
    const v16h*  Bt    = Bp + ((size_t)tn * kchunks) * 32 + lane;

    v8f acc = {};
    for (int c = 0; c < kchunks; ++c) {
        const float* Ak = Abase + (c << 5) + q;
        if (c + 1 < kchunks) {
            __builtin_prefetch(Abase + ((c + 1) << 5) + q, 0, 1); // global_prefetch_b8
        }
        float4 a0 = *(const float4*)(Ak);        // K = q..q+3
        float4 a1 = *(const float4*)(Ak + 4);    // K = q+4..q+7
        float4 a2 = *(const float4*)(Ak + 16);   // K = 16+q..16+q+3
        float4 a3 = *(const float4*)(Ak + 20);   // K = 16+q+4..16+q+7
        float av[16] = { a0.x, a0.y, a0.z, a0.w, a1.x, a1.y, a1.z, a1.w,
                         a2.x, a2.y, a2.z, a2.w, a3.x, a3.y, a3.z, a3.w };
        v16h a;
        #pragma unroll
        for (int h = 0; h < 16; ++h) a[h] = (_Float16)av[h];

        v16h b = Bt[(size_t)c * 32];

        acc = __builtin_amdgcn_wmma_f32_16x16x32_f16(
            /*neg_a=*/false, a, /*neg_b=*/false, b,
            /*c_mod=*/(short)0, acc, /*reuse_a=*/false, /*reuse_b=*/false);
    }

    // epilogue: C/D layout: VGPR rr -> M = rr + (lane>=16 ? 8 : 0), N = lane%16
    const int col = n0 + (lane & 15);
    if (col >= Nc) return;
    const float bv    = bias ? bias[col] : 0.0f;
    const int   rbase = m0 + ((lane & 16) ? 8 : 0);
    #pragma unroll
    for (int rr = 0; rr < 8; ++rr) {
        int orow = rbase + rr;
        if (orow < M) {
            float v = acc[rr] + bv;
            if (act == 1) v = fmaxf(v, 0.0f);
            else if (act == 2) v = 1.0f / (1.0f + __expf(-v));
            else if (act == 3) { v = fmaxf(v, 0.0f); v = 1.0f / (1.0f + __expf(-v)); }
            C[(size_t)orow * Nc + col] = v;
        }
    }
}

// ---------------------------------------------------------------------------
// GAT helper kernels
// ---------------------------------------------------------------------------
__global__ void rowdot_kernel(const float* __restrict__ h,
                              const float* __restrict__ a_src,
                              const float* __restrict__ a_dst,
                              float* __restrict__ ls, float* __restrict__ ld,
                              int n, int C) {
    int i = blockIdx.x * blockDim.x + threadIdx.x;
    if (i >= n) return;
    const float* hr = h + (size_t)i * C;
    float s0 = 0.f, s1 = 0.f;
    for (int c = 0; c < C; ++c) { float v = hr[c]; s0 += v * a_src[c]; s1 += v * a_dst[c]; }
    ls[i] = s0; ld[i] = s1;
}

__device__ __forceinline__ void edge_ends(const int* __restrict__ ei, int E,
                                          long long e, int& s, int& d) {
    if (e < E) { s = ei[e]; d = ei[(size_t)E + e]; }
    else       { s = d = (int)(e - E); }              // self loops
}

__global__ void edge_logit_kernel(const int* __restrict__ ei, int E, long long ET,
                                  const float* __restrict__ ls,
                                  const float* __restrict__ ld,
                                  float* __restrict__ logit,
                                  unsigned* __restrict__ mkey) {
    long long e = (long long)blockIdx.x * blockDim.x + threadIdx.x;
    if (e >= ET) return;
    int s, d; edge_ends(ei, E, e, s, d);
    float z = ls[s] + ld[d];
    if (z < 0.0f) z *= 0.2f;
    logit[e] = z;
    atomicMax(&mkey[d], ford(z));
}

__global__ void edge_exp_kernel(const int* __restrict__ ei, int E, long long ET,
                                const float* __restrict__ logit,
                                const unsigned* __restrict__ mkey,
                                float* __restrict__ eexp,
                                float* __restrict__ ssum) {
    long long e = (long long)blockIdx.x * blockDim.x + threadIdx.x;
    if (e >= ET) return;
    int s, d; edge_ends(ei, E, e, s, d);
    float m  = fdec(mkey[d]);
    float ex = __expf(logit[e] - m);
    eexp[e] = ex;
    atomicAdd(&ssum[d], ex);
}

__global__ void edge_scatter_kernel(const int* __restrict__ ei, int E, long long ET,
                                    const float* __restrict__ h,
                                    const float* __restrict__ eexp,
                                    const float* __restrict__ ssum,
                                    float* __restrict__ out, int C) {
    long long t = (long long)blockIdx.x * blockDim.x + threadIdx.x;
    long long total = ET * (long long)C;
    if (t >= total) return;
    long long e = t / C;
    int c = (int)(t - e * C);
    int s, d; edge_ends(ei, E, e, s, d);
    float alpha = eexp[e] / (ssum[d] + 1e-16f);
    atomicAdd(&out[(size_t)d * C + c], h[(size_t)s * C + c] * alpha);
}

__global__ void bias_relu_kernel(float* __restrict__ out,
                                 const float* __restrict__ b,
                                 long long n, int C) {
    long long i = (long long)blockIdx.x * blockDim.x + threadIdx.x;
    if (i >= n) return;
    out[i] = fmaxf(out[i] + b[i % C], 0.0f);
}

__global__ void fill_key_kernel(unsigned* __restrict__ p, float v, long long n) {
    long long i = (long long)blockIdx.x * blockDim.x + threadIdx.x;
    if (i < n) p[i] = ford(v);
}

__global__ void pool_max_kernel(const float* __restrict__ h,
                                const int* __restrict__ batch,
                                unsigned* __restrict__ gkey,
                                int n, int C) {
    long long t = (long long)blockIdx.x * blockDim.x + threadIdx.x;
    long long total = (long long)n * C;
    if (t >= total) return;
    int node = (int)(t / C);
    int c    = (int)(t - (long long)node * C);
    atomicMax(&gkey[(size_t)batch[node] * C + c], ford(h[t]));
}

__global__ void pool_decode_kernel(const unsigned* __restrict__ gkey,
                                   float* __restrict__ g, long long n) {
    long long i = (long long)blockIdx.x * blockDim.x + threadIdx.x;
    if (i < n) g[i] = fdec(gkey[i]);
}

// ---------------------------------------------------------------------------
// Host-side launcher
// ---------------------------------------------------------------------------
static inline long long cdiv_ll(long long a, long long b) { return (a + b - 1) / b; }

struct GemmCtx {
    _Float16* bp_cursor;
    hipStream_t stream;
};

static void launch_gemm(GemmCtx& ctx, const float* A, const float* B,
                        const float* bias, float* C, int M, int Nc, int K, int act) {
    const int kchunks = K >> 5;
    const int tilesN  = (Nc + 15) >> 4;
    _Float16* Bp = ctx.bp_cursor;
    const long long packElems = (long long)tilesN * kchunks * 32;     // v16h units
    ctx.bp_cursor += packElems * 16;

    pack_b_kernel<<<(unsigned)cdiv_ll(packElems, 256), 256, 0, ctx.stream>>>(B, Bp, K, Nc);

    long long tiles = (long long)cdiv_ll(M, 16) * tilesN;
    dim3 block(32, 4);
    dim3 grid((unsigned)cdiv_ll(tiles, 4));
    wmma_gemm_kernel<<<grid, block, 0, ctx.stream>>>(A, (const v16h*)Bp, bias, C, M, Nc, K, act);
}

extern "C" void kernel_launch(void* const* d_in, const int* in_sizes, int n_in,
                              void* d_out, int out_size, void* d_ws, size_t ws_size,
                              hipStream_t stream) {
    // ---- inputs (setup_inputs dict order) ----
    const float* x     = (const float*)d_in[0];
    const int*   ei    = (const int*)  d_in[1];
    const int*   batch = (const int*)  d_in[2];
    // d_in[3]=fp (unused by the reference), d_in[4]=num_graphs (scalar)
    const float* W1 = (const float*)d_in[5];  const float* as1 = (const float*)d_in[6];
    const float* ad1= (const float*)d_in[7];  const float* b1  = (const float*)d_in[8];
    const float* W2 = (const float*)d_in[9];  const float* as2 = (const float*)d_in[10];
    const float* ad2= (const float*)d_in[11]; const float* b2  = (const float*)d_in[12];
    const float* W3 = (const float*)d_in[13]; const float* as3 = (const float*)d_in[14];
    const float* ad3= (const float*)d_in[15]; const float* b3  = (const float*)d_in[16];
    const float* fcg1_w = (const float*)d_in[17]; const float* fcg1_b = (const float*)d_in[18];
    const float* fcg2_w = (const float*)d_in[19]; const float* fcg2_b = (const float*)d_in[20];
    const float* l1_w = (const float*)d_in[21]; const float* l1_b = (const float*)d_in[22];
    const float* l2_w = (const float*)d_in[23]; const float* l2_b = (const float*)d_in[24];
    const float* l3_w = (const float*)d_in[25]; const float* l3_b = (const float*)d_in[26];
    float* out = (float*)d_out;

    const int N  = in_sizes[0] / 32;       // 100000
    const int E  = in_sizes[1] / 2;        // 600000
    const int G  = in_sizes[3] / 16;       // 2000
    const long long ET = (long long)E + N; // edges + self loops

    // ---- workspace carve (floats) ----
    float* p = (float*)d_ws;
    float* nodeA = p; p += (size_t)N * 128;   // ping
    float* nodeB = p; p += (size_t)N * 128;   // pong
    float* hbuf  = p; p += (size_t)N * 128;   // h = in @ W
    float* ls    = p; p += N;
    float* ld    = p; p += N;
    float* logit = p; p += ET;
    float* eexp  = p; p += ET;
    unsigned* mkey = (unsigned*)p; p += N;
    float* ssum  = p; p += N;
    unsigned* gkey = (unsigned*)p; p += (size_t)G * 128;
    float* g   = p; p += (size_t)G * 128;
    float* g1  = p; p += (size_t)G * 1024;
    float* g2  = p; p += (size_t)G * 128;
    float* t1  = p; p += (size_t)G * 512;
    float* t2  = p; p += (size_t)G * 1024;
    _Float16* bpack = (_Float16*)p;           // packed-weight region (~1.8 MB)
    (void)ws_size; (void)n_in; (void)out_size;

    const int TB = 256;
    GemmCtx ctx{bpack, stream};

    // ---- one GAT layer ----
    auto gat_layer = [&](const float* in, int Cin, int Cout,
                         const float* W, const float* asrc, const float* adst,
                         const float* b, float* outbuf) {
        launch_gemm(ctx, in, W, nullptr, hbuf, N, Cout, Cin, 0);               // h = in@W (WMMA)
        rowdot_kernel<<<(unsigned)cdiv_ll(N, TB), TB, 0, stream>>>(hbuf, asrc, adst, ls, ld, N, Cout);
        (void)hipMemsetAsync(mkey, 0, (size_t)N * 4, stream);                  // key 0 == -inf sentinel
        edge_logit_kernel<<<(unsigned)cdiv_ll(ET, TB), TB, 0, stream>>>(ei, E, ET, ls, ld, logit, mkey);
        (void)hipMemsetAsync(ssum, 0, (size_t)N * 4, stream);
        edge_exp_kernel<<<(unsigned)cdiv_ll(ET, TB), TB, 0, stream>>>(ei, E, ET, logit, mkey, eexp, ssum);
        (void)hipMemsetAsync(outbuf, 0, (size_t)N * Cout * 4, stream);
        edge_scatter_kernel<<<(unsigned)cdiv_ll(ET * Cout, TB), TB, 0, stream>>>(ei, E, ET, hbuf, eexp, ssum, outbuf, Cout);
        bias_relu_kernel<<<(unsigned)cdiv_ll((long long)N * Cout, TB), TB, 0, stream>>>(outbuf, b, (long long)N * Cout, Cout);
    };

    gat_layer(x,     32,  32, W1, as1, ad1, b1, nodeA);
    gat_layer(nodeA, 32,  64, W2, as2, ad2, b2, nodeB);
    gat_layer(nodeB, 64, 128, W3, as3, ad3, b3, nodeA);

    // ---- global max pool -> g [G,128] ----
    {
        long long gn = (long long)G * 128;
        fill_key_kernel<<<(unsigned)cdiv_ll(gn, TB), TB, 0, stream>>>(gkey, -3.0e38f, gn);
        pool_max_kernel<<<(unsigned)cdiv_ll((long long)N * 128, TB), TB, 0, stream>>>(nodeA, batch, gkey, N, 128);
        pool_decode_kernel<<<(unsigned)cdiv_ll(gn, TB), TB, 0, stream>>>(gkey, g, gn);
    }

    // ---- MLP head (all WMMA) ----
    launch_gemm(ctx, g,  fcg1_w, fcg1_b, g1, G, 1024, 128,  1); // relu
    launch_gemm(ctx, g1, fcg2_w, fcg2_b, g2, G, 128,  1024, 2); // sigmoid
    launch_gemm(ctx, g2, l1_w,   l1_b,   t1, G, 512,  128,  1); // relu
    launch_gemm(ctx, t1, l2_w,   l2_b,   t2, G, 1024, 512,  1); // relu
    launch_gemm(ctx, t2, l3_w,   l3_b,  out, G, 1,    1024, 3); // relu -> sigmoid
}
